// NeighborhoodAttention3D_3865470566542
// MI455X (gfx1250) — matrix operand order repaired and checked by reading, not compiled, not run on hardware
//
#include <hip/hip_runtime.h>
#include <cstddef>

// ---------------------------------------------------------------------------
// NATTEN-3D on gfx1250:
//   convert(f32->bf16, weights transposed)  ->  QKV GEMM (TDM + bf16 WMMA)
//   ->  3x3x3 windowed attention (VALU)     ->  proj GEMM (TDM + bf16 WMMA)
// Sizes: tokens M=9216, C=384, NH=6, hd=64. GEMM1 N=1152, GEMM2 N=384, K=384.
// ---------------------------------------------------------------------------

typedef __attribute__((ext_vector_type(16))) __bf16   bf16x16;
typedef __attribute__((ext_vector_type(8)))  __bf16   bf16x8;
typedef __attribute__((ext_vector_type(8)))  float    v8f;
typedef __attribute__((ext_vector_type(4)))  unsigned u32x4;
typedef __attribute__((ext_vector_type(8)))  int      i32x8;
typedef __attribute__((ext_vector_type(4)))  int      i32x4;

#if __has_builtin(__builtin_amdgcn_tensor_load_to_lds) && \
    __has_builtin(__builtin_amdgcn_s_wait_tensorcnt)
#define USE_TDM 1
#else
#define USE_TDM 0
#endif

__device__ __forceinline__ __bf16 f32_to_bf16(float f) {
  unsigned u = __builtin_bit_cast(unsigned, f);
  unsigned r = (u + 0x7FFFu + ((u >> 16) & 1u)) >> 16;
  unsigned short s = (unsigned short)r;
  return __builtin_bit_cast(__bf16, s);
}

// Assemble a 16-element bf16 fragment from two 16-byte LDS chunks.
__device__ __forceinline__ bf16x16 frag16(const __bf16* p0, const __bf16* p1) {
  bf16x8 lo = *(const bf16x8*)p0;
  bf16x8 hi = *(const bf16x8*)p1;
  bf16x16 r;
#pragma unroll
  for (int i = 0; i < 8; ++i) { r[i] = lo[i]; r[i + 8] = hi[i]; }
  return r;
}

// ---------------------------------------------------------------------------
// Conversion kernels (run once per call; tiny vs GEMM traffic)
// ---------------------------------------------------------------------------
__global__ __launch_bounds__(256)
void cvt_bf16(const float* __restrict__ in, __bf16* __restrict__ out, int n) {
  int i = blockIdx.x * 256 + threadIdx.x;
  if (i < n) out[i] = f32_to_bf16(in[i]);
}

// in[K][N] (row-major) -> out[N][K] bf16 (row-major), i.e. transpose+convert.
__global__ __launch_bounds__(256)
void cvt_bf16_T(const float* __restrict__ in, __bf16* __restrict__ out, int N, int K) {
  int i = blockIdx.x * 256 + threadIdx.x;
  if (i < N * K) {
    int n = i / K, k = i % K;
    out[i] = f32_to_bf16(in[(size_t)k * N + n]);
  }
}

// ---------------------------------------------------------------------------
// GEMM: C[M x N](f32) = A[M x 384](bf16) * Bt[N x 384]^T(bf16) + bias[N]
// Block tile 128x64, 8 waves (4x2), each wave 32x32 via 4 WMMA accumulators.
// Tiles staged by the Tensor Data Mover with LDS padding (80B row stride).
// ---------------------------------------------------------------------------
#define TM 128
#define TN 64
#define TK 32
#define LDSA 40   // bf16 elems per LDS row: 32 data + 8 pad = 80 B (16B aligned)
#define LDSB 40

#if USE_TDM
// 2D tile load: `rows` rows of 32 bf16 (64B) with global row stride 384 elems,
// into LDS at lds_byte_off with +16B padding after every 64B (pad_interval=16DW,
// pad_amount=4DW) -> LDS row stride = 80B = LDSA/LDSB elements.
__device__ __forceinline__ void tdm_load_tile(const __bf16* gsrc,
                                              unsigned lds_byte_off, int rows) {
  unsigned long long ga = (unsigned long long)(size_t)gsrc;
  u32x4 g0;
  g0[0] = 1u;                                               // count=1
  g0[1] = lds_byte_off;                                     // lds_addr
  g0[2] = (unsigned)ga;                                     // global_addr[31:0]
  g0[3] = (unsigned)((ga >> 32) & 0x01FFFFFFull) | (2u << 30); // addr hi | type=2
  i32x8 g1;
  g1[0] = (int)((1u << 16)    // data_size = 2 bytes
              | (1u << 20)    // pad_enable
              | (3u << 22)    // pad_interval: 16 DWORDs (64 B)
              | (3u << 25));  // pad_amount:   4 DWORDs (16 B)
  g1[1] = (int)(32u << 16);                        // tensor_dim0 = 32
  g1[2] = (int)(((unsigned)rows & 0xFFFFu) << 16); // tensor_dim1 lo
  g1[3] = (int)(32u << 16);                        // tile_dim0 = 32
  g1[4] = (int)((unsigned)rows & 0xFFFFu);         // tile_dim1 = rows
  g1[5] = 384;                                     // tensor_dim0_stride = K
  g1[6] = 0;
  g1[7] = 0;
  i32x4 z4 = {};
#if __clang_major__ >= 23
  i32x8 z8 = {};
  __builtin_amdgcn_tensor_load_to_lds(g0, g1, z4, z4, z8, 0);
#else
  __builtin_amdgcn_tensor_load_to_lds(g0, g1, z4, z4, 0);
#endif
}
#endif

__global__ __launch_bounds__(256)
void gemm_bf16_wmma(const __bf16* __restrict__ A, const __bf16* __restrict__ Bt,
                    const float* __restrict__ bias, float* __restrict__ C, int N) {
  constexpr int K = 384;
  constexpr int NK = K / TK;   // 12 K-steps
  __shared__ __align__(16) __bf16 sA[2][TM * LDSA];
  __shared__ __align__(16) __bf16 sB[2][TN * LDSB];

  const int tid  = threadIdx.x;
  const int m0   = blockIdx.y * TM;
  const int n0   = blockIdx.x * TN;
  const int wave = tid >> 5;        // wave32
  const int lane = tid & 31;
  const int wm   = wave & 3;
  const int wn   = wave >> 2;
  const int lm   = lane & 15;
  const int kh   = lane >> 4;

  v8f acc[2][2] = {};

#if USE_TDM
  if (wave == 0) {
    tdm_load_tile(A  + (size_t)m0 * K, (unsigned)(size_t)(&sA[0][0]), TM);
    tdm_load_tile(Bt + (size_t)n0 * K, (unsigned)(size_t)(&sB[0][0]), TN);
  }
#endif

  for (int i = 0; i < NK; ++i) {
    const int buf = i & 1;
#if USE_TDM
    if (wave == 0) {
      if (i + 1 < NK) {
        // prefetch next tiles into the other buffer, then wait for current pair
        tdm_load_tile(A  + (size_t)m0 * K + (i + 1) * TK,
                      (unsigned)(size_t)(&sA[buf ^ 1][0]), TM);
        tdm_load_tile(Bt + (size_t)n0 * K + (i + 1) * TK,
                      (unsigned)(size_t)(&sB[buf ^ 1][0]), TN);
        __builtin_amdgcn_s_wait_tensorcnt(2);
      } else {
        __builtin_amdgcn_s_wait_tensorcnt(0);
      }
    }
    __syncthreads();
#else
    const int k0 = i * TK;
    for (int c = tid; c < (TM * TK) / 8; c += 256) {   // 512 x 16B chunks
      int row = c >> 2, col = (c & 3) * 8;
      *(bf16x8*)(&sA[buf][row * LDSA + col]) =
          *(const bf16x8*)(A + (size_t)(m0 + row) * K + k0 + col);
    }
    for (int c = tid; c < (TN * TK) / 8; c += 256) {   // 256 x 16B chunks
      int row = c >> 2, col = (c & 3) * 8;
      *(bf16x8*)(&sB[buf][row * LDSB + col]) =
          *(const bf16x8*)(Bt + (size_t)(n0 + row) * K + k0 + col);
    }
    __syncthreads();
#endif

    // Fragments per ISA VGPR layouts (CDNA5 ISA):
    //  A 16x32: lane(m=lm,kh): elems 0..7 -> K=kh*8.., elems 8..15 -> K=16+kh*8..
    //  B 32x16: lane(n=lm,kh): elems 0..7 -> K=kh*8.., elems 8..15 -> K=16+kh*8..
    const __bf16* sAb = &sA[buf][0];
    const __bf16* sBb = &sB[buf][0];
    bf16x16 afr[2], bfr[2];
#pragma unroll
    for (int mt = 0; mt < 2; ++mt) {
      const __bf16* base = sAb + (wm * 32 + mt * 16 + lm) * LDSA;
      afr[mt] = frag16(base + kh * 8, base + 16 + kh * 8);
    }
#pragma unroll
    for (int nt = 0; nt < 2; ++nt) {
      const __bf16* base = sBb + (wn * 32 + nt * 16 + lm) * LDSB;
      bfr[nt] = frag16(base + kh * 8, base + 16 + kh * 8);
    }
#pragma unroll
    for (int mt = 0; mt < 2; ++mt)
#pragma unroll
      for (int nt = 0; nt < 2; ++nt)
        acc[mt][nt] = __builtin_amdgcn_wmma_f32_16x16x32_bf16(
            false, afr[mt], false, bfr[nt], (short)0, acc[mt][nt], false, false);
    __syncthreads();
  }

  // Epilogue: C/D layout VGPR r -> M = r + 8*kh, N = lm (coalesced over lanes)
#pragma unroll
  for (int mt = 0; mt < 2; ++mt) {
#pragma unroll
    for (int nt = 0; nt < 2; ++nt) {
      const int gm = m0 + wm * 32 + mt * 16 + kh * 8;
      const int gn = n0 + wn * 32 + nt * 16 + lm;
      const float bv = bias[gn];
#pragma unroll
      for (int r = 0; r < 8; ++r)
        C[(size_t)(gm + r) * N + gn] = acc[mt][nt][r] + bv;
    }
  }
}

// ---------------------------------------------------------------------------
// Windowed attention: one 64-thread block per (token, head).
// qkv per token: [q(6*64) | k(6*64) | v(6*64)] = 1152 f32. Output bf16.
// ---------------------------------------------------------------------------
__global__ __launch_bounds__(64)
void natten_attn(const float* __restrict__ qkv, __bf16* __restrict__ out) {
  const int blk  = blockIdx.x;
  const int head = blk % 6;
  const int tok  = blk / 6;
  const int w = tok % 24;
  const int h = (tok / 24) % 24;
  const int d = (tok / 576) % 8;
  const int tbase = tok - (d * 576 + h * 24 + w);   // batch base token
  const int sd = min(max(d - 1, 0), 8 - 3);
  const int sh = min(max(h - 1, 0), 24 - 3);
  const int sw = min(max(w - 1, 0), 24 - 3);
  const int t = threadIdx.x;

  __shared__ float s_q[64];
  __shared__ float s_logit[27];

  s_q[t] = qkv[(size_t)tok * 1152 + head * 64 + t];
  __syncthreads();

  if (t < 27) {
    const int jd = t / 9, jh = (t / 3) % 3, jw = t % 3;
    const int tn = tbase + (sd + jd) * 576 + (sh + jh) * 24 + (sw + jw);
    const float* kp = qkv + (size_t)tn * 1152 + 384 + head * 64;
    float a = 0.f;
#pragma unroll 8
    for (int e = 0; e < 64; ++e) a += s_q[e] * kp[e];
    s_logit[t] = a * 0.125f;   // scale = hd^-0.5
  }
  __syncthreads();

  float mx = -1e30f;
#pragma unroll
  for (int j = 0; j < 27; ++j) mx = fmaxf(mx, s_logit[j]);
  float sum = 0.f;
#pragma unroll
  for (int j = 0; j < 27; ++j) sum += __expf(s_logit[j] - mx);

  float a = 0.f;
#pragma unroll
  for (int j = 0; j < 27; ++j) {
    const int jd = j / 9, jh = (j / 3) % 3, jw = j % 3;
    const int tn = tbase + (sd + jd) * 576 + (sh + jh) * 24 + (sw + jw);
    a += __expf(s_logit[j] - mx) * qkv[(size_t)tn * 1152 + 768 + head * 64 + t];
  }
  out[(size_t)tok * 384 + head * 64 + t] = f32_to_bf16(a / sum);
}

// ---------------------------------------------------------------------------
extern "C" void kernel_launch(void* const* d_in, const int* in_sizes, int n_in,
                              void* d_out, int out_size, void* d_ws, size_t ws_size,
                              hipStream_t stream) {
  (void)in_sizes; (void)n_in; (void)out_size; (void)ws_size;
  const float* x      = (const float*)d_in[0];
  const float* w_qkv  = (const float*)d_in[1];
  const float* b_qkv  = (const float*)d_in[2];
  const float* w_proj = (const float*)d_in[3];
  const float* b_proj = (const float*)d_in[4];
  float* out = (float*)d_out;

  // Workspace layout (256B aligned offsets)
  char* ws = (char*)d_ws;
  float*  qkv   = (float*) (ws + 0);                      // 9216*1152 f32 = 42.5 MB
  __bf16* xb    = (__bf16*)(ws + 42467328);               // 9216*384  bf16
  __bf16* att   = (__bf16*)(ws + 49545216);               // 9216*384  bf16
  __bf16* wqb_t = (__bf16*)(ws + 56623104);               // 1152*384  bf16 (w_qkv^T)
  __bf16* wpb_t = (__bf16*)(ws + 57507840);               // 384*384   bf16 (w_proj^T)

  // One-time (per call) conversions: x -> bf16, weights -> transposed bf16
  cvt_bf16  <<<dim3((9216 * 384 + 255) / 256), dim3(256), 0, stream>>>(x, xb, 9216 * 384);
  cvt_bf16_T<<<dim3((1152 * 384 + 255) / 256), dim3(256), 0, stream>>>(w_qkv,  wqb_t, 1152, 384);
  cvt_bf16_T<<<dim3((384  * 384 + 255) / 256), dim3(256), 0, stream>>>(w_proj, wpb_t, 384,  384);

  // GEMM1: qkv = x @ w_qkv + b_qkv   (M=9216, N=1152, K=384)
  gemm_bf16_wmma<<<dim3(1152 / TN, 9216 / TM), dim3(256), 0, stream>>>(
      xb, wqb_t, b_qkv, qkv, 1152);

  // windowed attention: 9216 tokens * 6 heads -> bf16
  natten_attn<<<dim3(9216 * 6), dim3(64), 0, stream>>>(qkv, att);

  // GEMM2: out = att @ w_proj + b_proj  (M=9216, N=384, K=384)
  gemm_bf16_wmma<<<dim3(384 / TN, 9216 / TM), dim3(256), 0, stream>>>(
      att, wpb_t, b_proj, out, 384);
}
